// ModifiedSVDTokenizer_55508157333919
// MI455X (gfx1250) — matrix operand order repaired and checked by reading, not compiled, not run on hardware
//
#include <hip/hip_runtime.h>
#include <hip/hip_bf16.h>
#include <math.h>

typedef __attribute__((ext_vector_type(16))) _Float16 v16h;
typedef __attribute__((ext_vector_type(8)))  float    v8f;
typedef __attribute__((ext_vector_type(4)))  int      v4i;

#define EPSV 1e-5f
#define LRELU(x) ((x) > 0.0f ? (x) : 0.01f * (x))

// ---------------- gfx1250 async global->LDS copy (ASYNCcnt path) ----------------
#if defined(__HIP_DEVICE_COMPILE__) && __has_builtin(__builtin_amdgcn_global_load_async_to_lds_b128)
#define HAVE_ASYNC_LDS 1
typedef __attribute__((address_space(1))) v4i gv4i;   // global (device) AS
typedef __attribute__((address_space(3))) v4i lv4i;   // LDS AS
#else
#define HAVE_ASYNC_LDS 0
#endif

__device__ __forceinline__ void async_copy16(const void* g, void* l) {
#if HAVE_ASYNC_LDS
    __builtin_amdgcn_global_load_async_to_lds_b128((gv4i*)g, (lv4i*)l, 0, 0);
#else
    *(uint4*)l = *(const uint4*)g;
#endif
}

__device__ __forceinline__ void async_wait() {
#if HAVE_ASYNC_LDS
#if __has_builtin(__builtin_amdgcn_s_wait_asynccnt)
    __builtin_amdgcn_s_wait_asynccnt(0);
#else
    asm volatile("s_wait_asynccnt 0" ::: "memory");
#endif
#endif
}

__device__ __forceinline__ float pos_enc_val(int s, int c) {
    // pe[s, 2i] = sin(s * exp(-2i*ln(1e4)/768)); pe[s,2i+1] = cos(same freq)
    int i2 = c & ~1;
    float freq = __expf(-(float)i2 * (9.210340371976184f / 768.0f));
    float ang = (float)s * freq;
    return (c & 1) ? __cosf(ang) : __sinf(ang);
}

// ---------------- small utility kernels ----------------
__global__ void zero_kernel(float* p, int n) {
    int i = blockIdx.x * blockDim.x + threadIdx.x;
    if (i < n) p[i] = 0.0f;
}

__global__ void f32_to_f16_kernel(const float* __restrict__ src, _Float16* __restrict__ dst, int n) {
    int i = blockIdx.x * blockDim.x + threadIdx.x;
    if (i < n) dst[i] = (_Float16)src[i];
}

// ---------------- conv3x3 + batch-stat accumulation ----------------
template<int CIN, int COUT, bool FUSE>
__global__ void conv3x3_kernel(const float* __restrict__ in,
                               const float* __restrict__ w,   // (COUT,CIN,3,3)
                               const float* __restrict__ bias,
                               const float* __restrict__ ac,  // [2*CIN] a then c (input BN)
                               float* __restrict__ out,
                               float* __restrict__ stats) {
    int p = blockIdx.x * blockDim.x + threadIdx.x;   // exactly 16*65536 threads
    int b = p >> 16;
    int hw = p & 65535;
    int h = hw >> 8, x = hw & 255;
    float acc[COUT];
#pragma unroll
    for (int co = 0; co < COUT; ++co) acc[co] = bias[co];
#pragma unroll
    for (int ci = 0; ci < CIN; ++ci) {
        const float* ip = in + (((size_t)(b * CIN + ci)) << 16);
        float sa = FUSE ? ac[ci] : 1.0f;
        float sc = FUSE ? ac[CIN + ci] : 0.0f;
#pragma unroll
        for (int ky = 0; ky < 3; ++ky) {
            int hh = h + ky - 1;
            if (hh < 0 || hh >= 256) continue;
#pragma unroll
            for (int kx = 0; kx < 3; ++kx) {
                int ww = x + kx - 1;
                if (ww < 0 || ww >= 256) continue;   // SAME zero pad (on transformed input)
                float v = ip[hh * 256 + ww];
                if (FUSE) { v = v * sa + sc; v = LRELU(v); }
#pragma unroll
                for (int co = 0; co < COUT; ++co)
                    acc[co] += w[((co * CIN + ci) * 3 + ky) * 3 + kx] * v;
            }
        }
    }
    __shared__ float ssum[COUT], ssq[COUT];
    if (threadIdx.x < COUT) { ssum[threadIdx.x] = 0.0f; ssq[threadIdx.x] = 0.0f; }
    __syncthreads();
#pragma unroll
    for (int co = 0; co < COUT; ++co) {
        out[(((size_t)(b * COUT + co)) << 16) + hw] = acc[co];
        atomicAdd(&ssum[co], acc[co]);
        atomicAdd(&ssq[co], acc[co] * acc[co]);
    }
    __syncthreads();
    if (threadIdx.x < COUT) {
        atomicAdd(&stats[threadIdx.x], ssum[threadIdx.x]);
        atomicAdd(&stats[COUT + threadIdx.x], ssq[threadIdx.x]);
    }
}

__global__ void bn_finalize_kernel(const float* __restrict__ stats,
                                   const float* __restrict__ g,
                                   const float* __restrict__ be,
                                   float* __restrict__ ac, int C, float invn) {
    int c = threadIdx.x;
    if (c < C) {
        float mean = stats[c] * invn;
        float var = stats[C + c] * invn - mean * mean;   // biased var
        float a = g[c] * rsqrtf(var + EPSV);
        ac[c] = a;
        ac[C + c] = be[c] - mean * a;
    }
}

// ---------------- pixel_unshuffle + BN + lrelu + f16 convert ----------------
__global__ void punshuffle_kernel(const float* __restrict__ y3,
                                  const float* __restrict__ ac,    // [24]
                                  _Float16* __restrict__ out) {
    int e = blockIdx.x * blockDim.x + threadIdx.x;    // exactly 16*4096*192
    int ci = e % 192;
    int t  = (e / 192) & 4095;
    int b  = e / (192 * 4096);
    int c  = ci >> 4;
    int r  = ci & 15;
    int dy = r >> 2, dx = r & 3;
    int ho = t >> 6, wo = t & 63;
    int h = ho * 4 + dy, w = wo * 4 + dx;
    float v = y3[(((size_t)(b * 12 + c)) << 16) + h * 256 + w];
    v = v * ac[c] + ac[12 + c];
    v = LRELU(v);
    out[e] = (_Float16)v;
}

// ---------------- GEMM 65536x192 @ 192x192 (1x1 conv), WMMA f16 ----------------
// block = 128 (4 waves); one block = 16 rows x all 192 cols; A staged in LDS once.
__global__ void gemm192_kernel(const _Float16* __restrict__ A,   // (65536,192)
                               const _Float16* __restrict__ Wt,  // (192,192) [co][ci]
                               _Float16* __restrict__ out,       // (65536,384)
                               int colOff) {
    __shared__ _Float16 sA[16][200];                  // 400B pitch, 16B aligned
    int tid = threadIdx.x;
    int row0 = blockIdx.x * 16;
    // stage 16x192 halfs = 384 16-byte chunks (24/row), async DMA to LDS
    for (int c0 = tid; c0 < 384; c0 += 128) {
        int r = c0 / 24, ch = c0 % 24;
        async_copy16(A + (size_t)(row0 + r) * 192 + ch * 8, &sA[r][ch * 8]);
    }
    async_wait();
    __syncthreads();
    int wid = tid >> 5, lane = tid & 31;
    int arow = lane & 15, kb  = (lane & 16) ? 16 : 0;
    int bn_  = lane & 15, kb2 = (lane & 16) ? 16 : 0;
    for (int cg = 0; cg < 3; ++cg) {
        int coltile = cg * 4 + wid;                   // 0..11
        const _Float16* Bb = Wt + (size_t)(coltile * 16 + bn_) * 192 + kb2;
        v8f c = {};
        for (int k0 = 0; k0 < 192; k0 += 32) {
            v16h a, bfr;
#pragma unroll
            for (int i = 0; i < 8; ++i)  a[i] = sA[arow][k0 + kb + i];
#pragma unroll
            for (int i = 8; i < 16; ++i) a[i] = sA[arow][k0 + kb + i];  // K = kb+8..kb+15
#pragma unroll
            for (int i = 0; i < 16; ++i) bfr[i] = Bb[k0 + i];
            c = __builtin_amdgcn_wmma_f32_16x16x32_f16(false, a, false, bfr, (short)0, c, false, false);
        }
        int n = lane & 15, mb = (lane & 16) ? 8 : 0;
#pragma unroll
        for (int v = 0; v < 8; ++v)
            out[(size_t)(row0 + v + mb) * 384 + colOff + coltile * 16 + n] = (_Float16)c[v];
    }
}

// ---------------- scorer: GEMM(384->128) + InstanceNorm + lrelu + dot(128)+relu ----------------
// block = 256 (8 waves = 8 coltiles of 16 -> 128 outputs); A staged in LDS once.
__global__ void scorer_kernel(const _Float16* __restrict__ raw,  // (65536,384)
                              const _Float16* __restrict__ w1,   // (128,384)
                              const float* __restrict__ b1,      // 128
                              const float* __restrict__ w2,      // 128
                              const float* __restrict__ b2,      // 1
                              float* __restrict__ sig) {         // 65536
    __shared__ _Float16 sA[16][392];                  // 784B pitch, 16B aligned
    __shared__ float sh[16][136];
    __shared__ float red[16][16];
    __shared__ float red2[16][16];
    __shared__ float smean[16], sinv[16];
    int tid = threadIdx.x;
    int row0 = blockIdx.x * 16;
    // stage 16x384 halfs = 768 chunks (48/row)
    for (int c0 = tid; c0 < 768; c0 += 256) {
        int r = c0 / 48, ch = c0 % 48;
        async_copy16(raw + (size_t)(row0 + r) * 384 + ch * 8, &sA[r][ch * 8]);
    }
    async_wait();
    __syncthreads();
    int wid = tid >> 5, lane = tid & 31;
    int coltile = wid;
    int arow = lane & 15, kb  = (lane & 16) ? 16 : 0;
    int bn_  = lane & 15, kb2 = (lane & 16) ? 16 : 0;
    const _Float16* Bb = w1 + (size_t)(coltile * 16 + bn_) * 384 + kb2;
    v8f c = {};
    for (int k0 = 0; k0 < 384; k0 += 32) {
        v16h a, bfr;
#pragma unroll
        for (int i = 0; i < 8; ++i)  a[i] = sA[arow][k0 + kb + i];
#pragma unroll
        for (int i = 8; i < 16; ++i) a[i] = sA[arow][k0 + kb + i];
#pragma unroll
        for (int i = 0; i < 16; ++i) bfr[i] = Bb[k0 + i];
        c = __builtin_amdgcn_wmma_f32_16x16x32_f16(false, a, false, bfr, (short)0, c, false, false);
    }
    int n = lane & 15, mb = (lane & 16) ? 8 : 0;
    int col = coltile * 16 + n;
    float bias = b1[col];
#pragma unroll
    for (int v = 0; v < 8; ++v) sh[v + mb][col] = c[v] + bias;
    __syncthreads();
    // per-token InstanceNorm over 128, lrelu, dot with w2, relu
    int t = tid >> 4, l = tid & 15;
    float s = 0.0f, sq = 0.0f;
#pragma unroll
    for (int j = 0; j < 8; ++j) { float x = sh[t][l * 8 + j]; s += x; sq += x * x; }
    red[t][l] = s; red2[t][l] = sq;
    __syncthreads();
    if (l == 0) {
        float S = 0.0f, Q = 0.0f;
        for (int j = 0; j < 16; ++j) { S += red[t][j]; Q += red2[t][j]; }
        float mean = S * (1.0f / 128.0f);
        float var = Q * (1.0f / 128.0f) - mean * mean;
        smean[t] = mean; sinv[t] = rsqrtf(var + EPSV);
    }
    __syncthreads();
    float mean = smean[t], inv = sinv[t];
    float acc = 0.0f;
#pragma unroll
    for (int j = 0; j < 8; ++j) {
        int cc = l * 8 + j;
        float x = (sh[t][cc] - mean) * inv;
        x = LRELU(x);
        acc += x * w2[cc];
    }
    __syncthreads();
    red[t][l] = acc;
    __syncthreads();
    if (l == 0) {
        float S = 0.0f;
        for (int j = 0; j < 16; ++j) S += red[t][j];
        S += b2[0];
        sig[row0 + t] = S > 0.0f ? S : 0.0f;
    }
}

// ---------------- per-batch bitonic sort (stable-argsort order) + cumsum + prefix mask ----------------
__global__ void sort_scan_kernel(const float* __restrict__ sig,
                                 int* __restrict__ gidx, float* __restrict__ gsig) {
    __shared__ float key[4096];
    __shared__ int   idx[4096];
    __shared__ float cs[4096];
    int b = blockIdx.x;
    int t = threadIdx.x;
#pragma unroll
    for (int e = 0; e < 4; ++e) {
        int i = t + e * 1024;
        key[i] = sig[b * 4096 + i];
        idx[i] = i;
    }
    __syncthreads();
    for (int size = 2; size <= 4096; size <<= 1) {
        for (int stride = size >> 1; stride > 0; stride >>= 1) {
#pragma unroll
            for (int e = 0; e < 4; ++e) {
                int i = t + e * 1024;
                int j = i ^ stride;
                if (j > i) {
                    float ki = key[i], kj = key[j];
                    int ii = idx[i], ij = idx[j];
                    bool up = ((i & size) == 0);
                    // (key, original index) lexicographic == stable argsort order
                    bool iLTj = (ki < kj) || (ki == kj && ii < ij);
                    bool dosw = up ? !iLTj : iLTj;
                    if (dosw) { key[i] = kj; key[j] = ki; idx[i] = ij; idx[j] = ii; }
                }
            }
            __syncthreads();
        }
    }
    // inclusive scan (Hillis-Steele)
#pragma unroll
    for (int e = 0; e < 4; ++e) cs[t + e * 1024] = key[t + e * 1024];
    __syncthreads();
    for (int d = 1; d < 4096; d <<= 1) {
        float tmp[4];
#pragma unroll
        for (int e = 0; e < 4; ++e) {
            int i = t + e * 1024;
            tmp[e] = cs[i] + ((i >= d) ? cs[i - d] : 0.0f);
        }
        __syncthreads();
#pragma unroll
        for (int e = 0; e < 4; ++e) cs[t + e * 1024] = tmp[e];
        __syncthreads();
    }
    float thresh = 0.9f * cs[4095];
#pragma unroll
    for (int e = 0; e < 4; ++e) {
        int i = t + e * 1024;
        bool keep = (cs[i] <= thresh);
        gidx[b * 4096 + i] = idx[i];
        gsig[b * 4096 + i] = keep ? key[i] : 0.0f;   // 0 => masked-out token contributes nothing
    }
}

// ---------------- final projection: gather sorted rows, GEMM 384->768, scale-by-sigma epilogue ----------------
// grid (256 rowtiles, 16 batches); block 256 (8 waves); A staged once, cg loop covers 48 coltiles.
// padded @ W == diag(gsig) * (raw_gather @ W)  -> scale rows in epilogue, A copy is a pure byte DMA.
__global__ void lp_gemm_kernel(const _Float16* __restrict__ raw,  // (16*4096,384)
                               const int* __restrict__ gidx,
                               const float* __restrict__ gsig,
                               const _Float16* __restrict__ lpw,  // (768,384)
                               const float* __restrict__ lpb,     // 768
                               float* __restrict__ out) {         // (16,4097,768)
    __shared__ _Float16 sA[16][392];
    __shared__ int   sGI[16];
    __shared__ float sGS[16];
    int b = blockIdx.y;
    int row0 = blockIdx.x * 16;
    int tid = threadIdx.x;
    if (tid < 16) {
        sGI[tid] = gidx[b * 4096 + row0 + tid];
        sGS[tid] = gsig[b * 4096 + row0 + tid];
    }
    __syncthreads();
    // gather 16 rows x 384 halfs = 768 chunks (48/row) via async DMA
    for (int c0 = tid; c0 < 768; c0 += 256) {
        int r = c0 / 48, ch = c0 % 48;
        const _Float16* src = raw + ((size_t)b * 4096 + sGI[r]) * 384;
        async_copy16(src + ch * 8, &sA[r][ch * 8]);
    }
    async_wait();
    __syncthreads();
    int wid = tid >> 5, lane = tid & 31;
    int arow = lane & 15, kb  = (lane & 16) ? 16 : 0;
    int bn_  = lane & 15, kb2 = (lane & 16) ? 16 : 0;
    for (int cg = 0; cg < 6; ++cg) {
        int coltile = cg * 8 + wid;                   // 0..47
        const _Float16* Bb = lpw + (size_t)(coltile * 16 + bn_) * 384 + kb2;
        v8f c = {};
        for (int k0 = 0; k0 < 384; k0 += 32) {
            v16h a, bfr;
#pragma unroll
            for (int i = 0; i < 8; ++i)  a[i] = sA[arow][k0 + kb + i];
#pragma unroll
            for (int i = 8; i < 16; ++i) a[i] = sA[arow][k0 + kb + i];
#pragma unroll
            for (int i = 0; i < 16; ++i) bfr[i] = Bb[k0 + i];
            c = __builtin_amdgcn_wmma_f32_16x16x32_f16(false, a, false, bfr, (short)0, c, false, false);
        }
        int n = lane & 15, mb = (lane & 16) ? 8 : 0;
        int col = coltile * 16 + n;
        float bias = lpb[col];
#pragma unroll
        for (int v = 0; v < 8; ++v) {
            int m = v + mb;
            int srow = 1 + row0 + m;                  // +1 for cls token row
            float val = sGS[m] * c[v] + bias + pos_enc_val(srow, col);
            out[((size_t)b * 4097 + srow) * 768 + col] = val;
        }
    }
}

__global__ void cls_pe_kernel(const float* __restrict__ cls, float* __restrict__ out) {
    int i = blockIdx.x * blockDim.x + threadIdx.x;
    if (i >= 16 * 768) return;
    int b = i / 768, c = i % 768;
    out[(size_t)b * 4097 * 768 + c] = cls[c] + pos_enc_val(0, c);
}

// ------------------------------------------------------------------
extern "C" void kernel_launch(void* const* d_in, const int* in_sizes, int n_in,
                              void* d_out, int out_size, void* d_ws, size_t ws_size,
                              hipStream_t stream) {
    const float* x = (const float*)d_in[0];
    // inputs: 1..12 = u path (w1,b1,g1,be1, w2,b2,g2,be2, w3,b3,g3,be3), 13..24 = v path
    const float* pu_w = (const float*)d_in[25];
    const float* pv_w = (const float*)d_in[26];
    const float* s_w1 = (const float*)d_in[27];
    const float* s_b1 = (const float*)d_in[28];
    const float* s_w2 = (const float*)d_in[29];
    const float* s_b2 = (const float*)d_in[30];
    const float* lp_w = (const float*)d_in[31];
    const float* lp_b = (const float*)d_in[32];
    const float* clsv = (const float*)d_in[33];
    float* out = (float*)d_out;

    char* Wp = (char*)d_ws;
    size_t off = 0;
    auto take = [&](size_t bytes) -> void* {
        void* p = Wp + off;
        off += (bytes + 255) & ~(size_t)255;
        return p;
    };
    float* y1 = (float*)take(16u * 4 * 65536 * 4);           // 16.8 MB
    float* y2 = (float*)take(16u * 6 * 65536 * 4);           // 25.2 MB
    _Float16* punsh = (_Float16*)y1;                         // aliases y1+y2 region (25.2 MB fits in 42 MB)
    float* y3 = (float*)take(16ull * 12 * 65536 * 4);        // 50.3 MB
    _Float16* raw = (_Float16*)take(16ull * 4096 * 384 * 2); // 50.3 MB
    _Float16* puw16 = (_Float16*)take(192 * 192 * 2);
    _Float16* pvw16 = (_Float16*)take(192 * 192 * 2);
    _Float16* sw116 = (_Float16*)take(128 * 384 * 2);
    _Float16* lpw16 = (_Float16*)take(768 * 384 * 2);
    float* st  = (float*)take(256);
    float* ac1 = (float*)take(256);
    float* ac2 = (float*)take(256);
    float* ac3 = (float*)take(256);
    float* sig = (float*)take(65536 * 4);
    int*   gidx = (int*)take(65536 * 4);
    float* gsig = (float*)take(65536 * 4);

    const float INVN = 1.0f / (16.0f * 256.0f * 256.0f);

    // weight conversions to f16
    f32_to_f16_kernel<<<(36864 + 255) / 256, 256, 0, stream>>>(pu_w, puw16, 36864);
    f32_to_f16_kernel<<<(36864 + 255) / 256, 256, 0, stream>>>(pv_w, pvw16, 36864);
    f32_to_f16_kernel<<<(49152 + 255) / 256, 256, 0, stream>>>(s_w1, sw116, 49152);
    f32_to_f16_kernel<<<(294912 + 255) / 256, 256, 0, stream>>>(lp_w, lpw16, 294912);

    for (int path = 0; path < 2; ++path) {
        int base = 1 + path * 12;
        const float* w1 = (const float*)d_in[base + 0];
        const float* b1 = (const float*)d_in[base + 1];
        const float* g1 = (const float*)d_in[base + 2];
        const float* be1 = (const float*)d_in[base + 3];
        const float* w2 = (const float*)d_in[base + 4];
        const float* b2 = (const float*)d_in[base + 5];
        const float* g2 = (const float*)d_in[base + 6];
        const float* be2 = (const float*)d_in[base + 7];
        const float* w3 = (const float*)d_in[base + 8];
        const float* b3 = (const float*)d_in[base + 9];
        const float* g3 = (const float*)d_in[base + 10];
        const float* be3 = (const float*)d_in[base + 11];

        zero_kernel<<<1, 32, 0, stream>>>(st, 8);
        conv3x3_kernel<3, 4, false><<<4096, 256, 0, stream>>>(x, w1, b1, ac1, y1, st);
        bn_finalize_kernel<<<1, 32, 0, stream>>>(st, g1, be1, ac1, 4, INVN);

        zero_kernel<<<1, 32, 0, stream>>>(st, 12);
        conv3x3_kernel<4, 6, true><<<4096, 256, 0, stream>>>(y1, w2, b2, ac1, y2, st);
        bn_finalize_kernel<<<1, 32, 0, stream>>>(st, g2, be2, ac2, 6, INVN);

        zero_kernel<<<1, 32, 0, stream>>>(st, 24);
        conv3x3_kernel<6, 12, true><<<4096, 256, 0, stream>>>(y2, w3, b3, ac2, y3, st);
        bn_finalize_kernel<<<1, 32, 0, stream>>>(st, g3, be3, ac3, 12, INVN);

        punshuffle_kernel<<<49152, 256, 0, stream>>>(y3, ac3, punsh);
        gemm192_kernel<<<4096, 128, 0, stream>>>(punsh,
                                                 path == 0 ? puw16 : pvw16,
                                                 raw, path == 0 ? 0 : 192);
    }

    scorer_kernel<<<4096, 256, 0, stream>>>(raw, sw116, s_b1, s_w2, s_b2, sig);
    sort_scan_kernel<<<16, 1024, 0, stream>>>(sig, gidx, gsig);
    lp_gemm_kernel<<<dim3(256, 16), 256, 0, stream>>>(raw, gidx, gsig, lpw16, lp_b, out);
    cls_pe_kernel<<<(16 * 768 + 255) / 256, 256, 0, stream>>>(clsv, out);
}
